// ProbAttention_81801947119816
// MI455X (gfx1250) — matrix operand order repaired and checked by reading, not compiled, not run on hardware
//
#include <hip/hip_runtime.h>
#include <hip/hip_bf16.h>

// ---- problem constants (from reference) ----
#define BB 4
#define HH 8
#define LL 2048
#define DD 64
#define NS 40        // sample_k
#define NT 40        // n_top
#define NTP 48       // n_top padded to 3x16 row tiles
#define BKV 64       // key/value block streamed per iteration
#define NBH (BB*HH)  // 32
#define NBLK (LL/BKV)

typedef __attribute__((ext_vector_type(16))) __bf16 v16bf;
typedef __attribute__((ext_vector_type(8)))  float  v8f;

union V16BF { v16bf v; __bf16 e[16]; };
union V8F   { v8f   v; float  e[8];  };

// LDS pitches (pad to dodge bank conflicts; keep 16B-divisible byte pitch for b128)
#define QP 72   // bf16 pitch for Q tile           (144 B)
#define KP 72   // bf16 pitch for K/V tiles        (144 B)
#define PP 72   // bf16 pitch for P (exp scores)   (144 B)
#define SP 65   // f32 pitch for raw scores

// ---- shared-memory arena layout for the attention kernel (single var => offset 0) ----
#define KS_OFF(b)  ((b) * (BKV * KP * 2))                 // 2 x 9216
#define VS_OFF(b)  (2 * (BKV * KP * 2) + (b) * (BKV * KP * 2))
#define QS_OFF     (4 * (BKV * KP * 2))                   // 36864
#define PS_OFF     (QS_OFF + NTP * QP * 2)                // +6912 = 43776
#define ST_OFF     (PS_OFF + NTP * PP * 2)                // +6912 = 50688
#define RM_OFF     (ST_OFF + NTP * SP * 4)                // +12480 = 63168
#define RS_OFF     (RM_OFF + NTP * 4)                     // 63360
#define CR_OFF     (RS_OFF + NTP * 4)                     // 63552
#define MT_OFF     (CR_OFF + NTP * 4)                     // 63744
#define SMEM_BYTES (MT_OFF + NTP * 4)                     // 63936

// ---------------------------------------------------------------
// Kernel 0: convert K and V to bf16 in workspace (enables async DMA)
// ---------------------------------------------------------------
__global__ void pa_tobf16(const float* __restrict__ k, const float* __restrict__ v,
                          __bf16* __restrict__ kbf, __bf16* __restrict__ vbf) {
    int gid = blockIdx.x * blockDim.x + threadIdx.x;
    if (gid >= NBH * LL * DD) return;
    kbf[gid] = (__bf16)k[gid];
    vbf[gid] = (__bf16)v[gid];
}

// ---------------------------------------------------------------
// Kernel 1: M[b,h,l] = max_s(q_l . k_idx[l,s]) - sum_s(...)/L
// ---------------------------------------------------------------
__global__ void pa_sample_score(const float* __restrict__ q,
                                const float* __restrict__ k,
                                const int*   __restrict__ idx,
                                float* __restrict__ Mout) {
    int gid = blockIdx.x * blockDim.x + threadIdx.x;   // 0 .. NBH*LL-1
    if (gid >= NBH * LL) return;
    int l  = gid & (LL - 1);
    int bh = gid >> 11;
    const float* qrow  = q + (size_t)gid * DD;
    const float* kbase = k + (size_t)bh * LL * DD;

    float qreg[DD];
#pragma unroll
    for (int d = 0; d < DD; ++d) qreg[d] = qrow[d];

    float mx = -3.4e38f, sm = 0.f;
    for (int s = 0; s < NS; ++s) {
        int ki = idx[l * NS + s];
        const float* krow = kbase + (size_t)ki * DD;
        float dot = 0.f;
#pragma unroll
        for (int d = 0; d < DD; ++d) dot += qreg[d] * krow[d];
        mx = fmaxf(mx, dot);
        sm += dot;
    }
    Mout[gid] = mx - sm * (1.0f / (float)LL);
}

// ---------------------------------------------------------------
// Kernel 2: per-(b,h) top-40 indices of M (iterative argmax)
// ---------------------------------------------------------------
__global__ void pa_topk(const float* __restrict__ M, int* __restrict__ mtop) {
    __shared__ float vals[LL];
    __shared__ float redv[256];
    __shared__ int   redi[256];
    int bh  = blockIdx.x;
    int tid = threadIdx.x;

    for (int i = tid; i < LL; i += 256) vals[i] = M[(size_t)bh * LL + i];
    __syncthreads();

    for (int t = 0; t < NT; ++t) {
        float bv = -3.4e38f; int bi = LL;
        for (int i = tid; i < LL; i += 256) {
            float v = vals[i];
            if (v > bv) { bv = v; bi = i; }
        }
        redv[tid] = bv; redi[tid] = bi;
        __syncthreads();
        for (int s = 128; s > 0; s >>= 1) {
            if (tid < s) {
                if (redv[tid + s] > redv[tid] ||
                    (redv[tid + s] == redv[tid] && redi[tid + s] < redi[tid])) {
                    redv[tid] = redv[tid + s];
                    redi[tid] = redi[tid + s];
                }
            }
            __syncthreads();
        }
        if (tid == 0) {
            mtop[bh * NT + t] = redi[0];
            vals[redi[0]] = -3.4e38f;
        }
        __syncthreads();
    }
}

// ---------------------------------------------------------------
// Kernel 3: V_mean[b,h,d] = mean over L
// ---------------------------------------------------------------
__global__ void pa_vmean(const float* __restrict__ v, float* __restrict__ vmean) {
    int bh = blockIdx.x;
    int d  = threadIdx.x;     // 64 threads
    const float* vb = v + (size_t)bh * LL * DD;
    float s = 0.f;
    for (int l = 0; l < LL; ++l) s += vb[(size_t)l * DD + d];
    vmean[bh * DD + d] = s * (1.0f / (float)LL);
}

// ---------------------------------------------------------------
// Kernel 4: broadcast V_mean into every output row
// ---------------------------------------------------------------
__global__ void pa_fill(const float* __restrict__ vmean, float* __restrict__ out) {
    int gid = blockIdx.x * blockDim.x + threadIdx.x;
    if (gid >= NBH * LL * DD) return;
    out[gid] = vmean[(gid >> 17) * DD + (gid & 63)];   // LL*DD = 2^17
}

// ---------------------------------------------------------------
// Kernel 5: fused top-40 attention, bf16 WMMA + online softmax,
// K/V double-buffered in LDS via GLOBAL_LOAD_ASYNC_TO_LDS_B128.
// 1 block per (b,h), 128 threads = 4 waves; wave w owns col strip w.
// ---------------------------------------------------------------
__global__ void __launch_bounds__(128)
pa_attn_topk(const float* __restrict__ q,
             const __bf16* __restrict__ kbf,
             const __bf16* __restrict__ vbf,
             const int*   __restrict__ mtop,
             float* __restrict__ out) {
    __shared__ __align__(16) char smem[SMEM_BYTES];

    __bf16* qs     = (__bf16*)(smem + QS_OFF);
    __bf16* ps     = (__bf16*)(smem + PS_OFF);
    float*  st     = (float*) (smem + ST_OFF);
    float*  rowmax = (float*) (smem + RM_OFF);
    float*  rowsum = (float*) (smem + RS_OFF);
    float*  corr   = (float*) (smem + CR_OFF);
    int*    mtop_s = (int*)   (smem + MT_OFF);

    const int bh   = blockIdx.x;
    const int tid  = threadIdx.x;
    const int wave = tid >> 5;
    const int lane = tid & 31;
    const int lo16 = lane & 15;
    const int half = lane >> 4;

    const float* qb = q + (size_t)bh * LL * DD;
    const char*  kg = (const char*)(kbf + (size_t)bh * LL * DD);
    const char*  vg = (const char*)(vbf + (size_t)bh * LL * DD);

    // Issue one 64x64 bf16 K block + V block into LDS buffer `buf` asynchronously.
    // 512 16B chunks per tile; 128 threads -> 4 chunks each -> 8 async ops per wave.
    auto issue_block = [&](int kb0, int buf) {
        const char* kblk = kg + (size_t)kb0 * DD * 2;   // contiguous 8KB
        const char* vblk = vg + (size_t)kb0 * DD * 2;
        for (int c = tid; c < (BKV * DD * 2) / 16; c += 128) {
            unsigned lds_k = (unsigned)(KS_OFF(buf) + (c >> 3) * (KP * 2) + (c & 7) * 16);
            unsigned lds_v = (unsigned)(VS_OFF(buf) + (c >> 3) * (KP * 2) + (c & 7) * 16);
            unsigned long long ga_k = (unsigned long long)(kblk + (size_t)c * 16);
            unsigned long long ga_v = (unsigned long long)(vblk + (size_t)c * 16);
            asm volatile("global_load_async_to_lds_b128 %0, %1, off"
                         :: "v"(lds_k), "v"(ga_k) : "memory");
            asm volatile("global_load_async_to_lds_b128 %0, %1, off"
                         :: "v"(lds_v), "v"(ga_v) : "memory");
        }
    };

    if (tid < NTP) {
        mtop_s[tid] = (tid < NT) ? mtop[bh * NT + tid] : -1;
        rowmax[tid] = -3.4e38f;
        rowsum[tid] = 0.f;
    }

    issue_block(0, 0);          // prefetch first K/V block while we gather Q

    __syncthreads();            // mtop_s visible

    // gather Q_reduce (48x64, padded rows zero) as bf16
    for (int i = tid; i < NTP * DD; i += 128) {
        int r = i >> 6, c = i & 63;
        float val = (mtop_s[r] >= 0) ? qb[(size_t)mtop_s[r] * DD + c] : 0.f;
        qs[r * QP + c] = (__bf16)val;
    }

    V8F acc[3];
#pragma unroll
    for (int t = 0; t < 3; ++t)
#pragma unroll
        for (int i = 0; i < 8; ++i) acc[t].e[i] = 0.f;

    const float scale = 0.125f;   // 1/sqrt(64)

    for (int it = 0; it < NBLK; ++it) {
        const int buf = it & 1;
        if (it + 1 < NBLK) {
            issue_block((it + 1) * BKV, buf ^ 1);
            // 16 async ops outstanding; oldest 8 (current block) retire in order
            asm volatile("s_wait_asynccnt 0x8" ::: "memory");
        } else {
            asm volatile("s_wait_asynccnt 0x0" ::: "memory");
        }
        __syncthreads();        // current buffer's K/V visible block-wide

        const __bf16* ksb = (const __bf16*)(smem + KS_OFF(buf));
        const __bf16* vsb = (const __bf16*)(smem + VS_OFF(buf));

        // ---- S = Q (48x64) x K^T (64x16 strip per wave), scaled ----
#pragma unroll
        for (int rt = 0; rt < 3; ++rt) {
            V8F c;
#pragma unroll
            for (int i = 0; i < 8; ++i) c.e[i] = 0.f;
#pragma unroll
            for (int k0 = 0; k0 < DD; k0 += 32) {
                V16BF a, b;
#pragma unroll
                for (int j = 0; j < 16; ++j) {
                    int kk = half * 16 + j;       // A layout (contiguous k, matches B)
                    a.e[j] = qs[(rt * 16 + lo16) * QP + k0 + kk];
                    b.e[j] = ksb[(wave * 16 + lo16) * KP + k0 + half * 16 + j]; // B layout
                }
                c.v = __builtin_amdgcn_wmma_f32_16x16x32_bf16(
                        false, a.v, false, b.v, (short)0, c.v, false, false);
            }
#pragma unroll
            for (int i = 0; i < 8; ++i) {
                int m = rt * 16 + half * 8 + i;
                st[m * SP + wave * 16 + lo16] = c.e[i] * scale;
            }
        }
        __syncthreads();

        // ---- online softmax over this 64-key block ----
        if (tid < NTP) {
            int r = tid;
            float m_old = rowmax[r];
            float bm = -3.4e38f;
#pragma unroll 8
            for (int c = 0; c < BKV; ++c) bm = fmaxf(bm, st[r * SP + c]);
            float m_new = fmaxf(m_old, bm);
            float cr = __expf(m_old - m_new);
            float s = 0.f;
#pragma unroll 8
            for (int c = 0; c < BKV; ++c) {
                float e = __expf(st[r * SP + c] - m_new);
                ps[r * PP + c] = (__bf16)e;
                s += e;
            }
            rowsum[r] = rowsum[r] * cr + s;
            rowmax[r] = m_new;
            corr[r]   = cr;
        }
        __syncthreads();

        // ---- rescale accumulators, then acc += P (48x64) x V (64x16 strip) ----
#pragma unroll
        for (int rt = 0; rt < 3; ++rt) {
#pragma unroll
            for (int i = 0; i < 8; ++i)
                acc[rt].e[i] *= corr[rt * 16 + half * 8 + i];
#pragma unroll
            for (int k0 = 0; k0 < BKV; k0 += 32) {
                V16BF a, b;
#pragma unroll
                for (int j = 0; j < 16; ++j) {
                    int kk = half * 16 + j;
                    a.e[j] = ps[(rt * 16 + lo16) * PP + k0 + kk];
                    b.e[j] = vsb[(k0 + half * 16 + j) * KP + wave * 16 + lo16];
                }
                acc[rt].v = __builtin_amdgcn_wmma_f32_16x16x32_bf16(
                        false, a.v, false, b.v, (short)0, acc[rt].v, false, false);
            }
        }
        __syncthreads();        // all reads of `buf` + ps done before overwrite
    }

    // ---- scatter normalized rows into output at top-k positions ----
#pragma unroll
    for (int rt = 0; rt < 3; ++rt) {
#pragma unroll
        for (int i = 0; i < 8; ++i) {
            int r = rt * 16 + half * 8 + i;
            int key = mtop_s[r];
            if (key >= 0) {
                int d = wave * 16 + lo16;
                out[((size_t)bh * LL + key) * DD + d] = acc[rt].e[i] / rowsum[r];
            }
        }
    }
}

// ---------------------------------------------------------------
extern "C" void kernel_launch(void* const* d_in, const int* in_sizes, int n_in,
                              void* d_out, int out_size, void* d_ws, size_t ws_size,
                              hipStream_t stream) {
    const float* q   = (const float*)d_in[0];
    const float* k   = (const float*)d_in[1];
    const float* v   = (const float*)d_in[2];
    const int*   idx = (const int*)d_in[3];
    float*       out = (float*)d_out;

    // workspace layout
    char* ws = (char*)d_ws;
    float*  Mws   = (float*)(ws);                          // NBH*LL f32   = 256 KB
    int*    mtop  = (int*)  (ws + 262144);                 // NBH*NT ints  (pad 8 KB)
    float*  vmean = (float*)(ws + 270336);                 // NBH*DD f32   (pad 8 KB)
    __bf16* kbf   = (__bf16*)(ws + 278528);                // 8 MB
    __bf16* vbf   = (__bf16*)(ws + 278528 + (size_t)NBH * LL * DD * 2);  // 8 MB

    (void)in_sizes; (void)n_in; (void)out_size; (void)ws_size;

    pa_tobf16<<<dim3((NBH * LL * DD + 255) / 256), dim3(256), 0, stream>>>(k, v, kbf, vbf);
    pa_sample_score<<<dim3((NBH * LL + 255) / 256), dim3(256), 0, stream>>>(q, k, idx, Mws);
    pa_topk<<<dim3(NBH), dim3(256), 0, stream>>>(Mws, mtop);
    pa_vmean<<<dim3(NBH), dim3(DD), 0, stream>>>(v, vmean);
    pa_fill<<<dim3((NBH * LL * DD + 255) / 256), dim3(256), 0, stream>>>(vmean, out);
    pa_attn_topk<<<dim3(NBH), dim3(128), 0, stream>>>(q, kbf, vbf, mtop, out);
}
